// MSDeformAttn1D_33758442946929
// MI455X (gfx1250) — matrix operand order repaired and checked by reading, not compiled, not run on hardware
//
#include <hip/hip_runtime.h>
#include <hip/hip_bf16.h>
#include <math.h>

#define DIMC   1024
#define NH     16
#define DHD    64
#define NL     3
#define NKP    4
#define BB     2
#define TT     2048
#define MAXOFF 0.25f

typedef __attribute__((ext_vector_type(16))) _Float16 v16h;
typedef __attribute__((ext_vector_type(8)))  _Float16 v8h;
typedef __attribute__((ext_vector_type(8)))  float    v8f;
typedef __attribute__((ext_vector_type(4)))  int      v4i;

#if defined(__gfx1250__) && __has_builtin(__builtin_amdgcn_global_load_async_to_lds_b128) && __has_builtin(__builtin_amdgcn_s_wait_asynccnt)
#define USE_ASYNC_LDS 1
typedef const __attribute__((address_space(1))) v4i* gv4i_cp;
typedef __attribute__((address_space(1))) v4i*       gv4i_p;
typedef __attribute__((address_space(3))) v4i*       lv4i_p;
#else
#define USE_ASYNC_LDS 0
#endif

// ----------------------------- elementwise helpers -----------------------------

__global__ void cvt_f32_to_f16_k(const float* __restrict__ in, _Float16* __restrict__ out, int n) {
  int i = blockIdx.x * blockDim.x + threadIdx.x;
  if (i < n) out[i] = (_Float16)in[i];
}

__global__ void mask_to_int_k(const unsigned char* __restrict__ m, int* __restrict__ o, int n) {
  int i = blockIdx.x * blockDim.x + threadIdx.x;
  if (i < n) o[i] = m[i] ? 1 : 0;
}

// Wt[n*K + k] = (f16) W[k*N + n]   (W is [K,N] row-major fp32)
__global__ void transpose_cvt_k(const float* __restrict__ W, _Float16* __restrict__ Wt, int K, int N) {
  int i = blockIdx.x * blockDim.x + threadIdx.x;
  if (i < K * N) {
    int n = i % N;
    int k = i / N;
    Wt[(size_t)n * K + k] = (_Float16)W[i];
  }
}

// masked average pool by 2 along T (Tin even here at every level)
__global__ void downsample_k(const float* __restrict__ fin, const int* __restrict__ mi,
                             float* __restrict__ fout, int* __restrict__ mo,
                             int Tin, int Tout) {
  int i = blockIdx.x * blockDim.x + threadIdx.x;
  int total = BB * Tout * DIMC;
  if (i >= total) return;
  int c = i % DIMC;
  int j = (i / DIMC) % Tout;
  int b = i / (DIMC * Tout);
  int m0 = mi[b * Tin + 2 * j];
  int m1 = mi[b * Tin + 2 * j + 1];
  float x0 = fin[((size_t)b * Tin + 2 * j) * DIMC + c];
  float x1 = fin[((size_t)b * Tin + 2 * j + 1) * DIMC + c];
  float denom = fmaxf((float)(m0 + m1), 1.0f);
  fout[((size_t)b * Tout + j) * DIMC + c] = (x0 * (float)m0 + x1 * (float)m1) / denom;
  if (c == 0) mo[b * Tout + j] = m0 | m1;
}

// ----------------------------- WMMA GEMM -----------------------------
// C[M,N] = A[M,K]f16 @ Bt[N,K]f16^T + bias.
// Block tile: 128 (M) x 64 (N), 8 waves; wave w owns rows [w*16, w*16+16) and all
// four 16-col subtiles (4 accumulators, 4 WMMAs per 32-K step, A fragment reused 4x).
// B chunk (64 cols x 32 K) staged in LDS, double buffered, shared by all waves;
// staged with global_load_async_to_lds_b128 when available (ASYNCcnt pipeline).
// Epilogues: 0 f32+bias | 1 tanh*0.25 | 2 f16 | 3 f32*rowMask.

__global__ __launch_bounds__(256)
void gemm_wmma_f16_k(const _Float16* __restrict__ A, const _Float16* __restrict__ Bt,
                     const float* __restrict__ bias,
                     float* __restrict__ outF, _Float16* __restrict__ outH,
                     const unsigned char* __restrict__ rowMask,
                     int M, int N, int Kd, int mode) {
  __shared__ __align__(64) _Float16 sB[2][64 * 32];   // [buf][col*32 + k]

  const int tid  = threadIdx.x;
  const int lane = tid & 31;
  const int wave = tid >> 5;
  const int nbn  = N >> 6;                      // N blocks of 64
  const int bm   = blockIdx.x / nbn;
  const int bn   = blockIdx.x - bm * nbn;
  const int n0   = bn * 64;
  const int tmRow = bm * 128 + wave * 16;

  const int halfId = lane >> 4;                 // 0: lanes 0-15, 1: lanes 16-31
  const int l15    = lane & 15;
  const int kb2    = halfId ? 16 : 0;
  // A fragment (16x32 f16): lane holds row l15, contiguous K chunk [kb2..kb2+15]
  const _Float16* aBase = A + (size_t)(tmRow + l15) * Kd + (halfId ? 16 : 0);

  // staging: each of 256 threads moves 16B per K step (64 cols x 64B)
  const int sc   = tid >> 2;                    // 0..63 : column within N block
  const int sseg = tid & 3;                     // 16B segment of the 32-half row chunk
  const _Float16* gB = Bt + (size_t)(n0 + sc) * Kd + sseg * 8;
  _Float16* lB[2] = { &sB[0][sc * 32 + sseg * 8], &sB[1][sc * 32 + sseg * 8] };

  v8f acc0 = {}, acc1 = {}, acc2 = {}, acc3 = {};

  auto stage = [&](int k0, int buf) {
#if USE_ASYNC_LDS
    __builtin_amdgcn_global_load_async_to_lds_b128(
        (gv4i_p)(gB + k0), (lv4i_p)lB[buf], 0, 0);
#else
    *(v8h*)lB[buf] = *(const v8h*)(gB + k0);
#endif
  };

  stage(0, 0);
  const int nsteps = Kd >> 5;
  for (int i = 0; i < nsteps; ++i) {
    const int k0 = i << 5;
#if USE_ASYNC_LDS
    __builtin_amdgcn_s_wait_asynccnt(0);
#endif
    __syncthreads();                            // buf[i&1] ready; prev reads of buf[(i+1)&1] done
    if (i + 1 < nsteps) stage(k0 + 32, (i + 1) & 1);

    union { v16h v; v8h h[2]; } af;
    af.h[0] = *(const v8h*)(aBase + k0);
    af.h[1] = *(const v8h*)(aBase + k0 + 8);
    if (i + 1 < nsteps) {
      __builtin_prefetch(aBase + k0 + 32, 0, 1);    // global_prefetch_b8
    }
    const _Float16* lbuf = sB[i & 1];
    // B fragment (32x16 f16): lane holds col l15(+16s), contiguous K chunk [kb2..kb2+15]
    v16h b0 = *(const v16h*)(lbuf + (0 * 16 + l15) * 32 + kb2);
    v16h b1 = *(const v16h*)(lbuf + (1 * 16 + l15) * 32 + kb2);
    v16h b2 = *(const v16h*)(lbuf + (2 * 16 + l15) * 32 + kb2);
    v16h b3 = *(const v16h*)(lbuf + (3 * 16 + l15) * 32 + kb2);
    acc0 = __builtin_amdgcn_wmma_f32_16x16x32_f16(false, af.v, false, b0, (short)0, acc0, false, false);
    acc1 = __builtin_amdgcn_wmma_f32_16x16x32_f16(false, af.v, false, b1, (short)0, acc1, false, false);
    acc2 = __builtin_amdgcn_wmma_f32_16x16x32_f16(false, af.v, false, b2, (short)0, acc2, false, false);
    acc3 = __builtin_amdgcn_wmma_f32_16x16x32_f16(false, af.v, false, b3, (short)0, acc3, false, false);
  }

  // C/D layout: lanes 0-15 -> N=lane, VGPR i -> M=i ; lanes 16-31 -> N=lane-16, M=8+i
  v8f accs[4] = {acc0, acc1, acc2, acc3};
  #pragma unroll
  for (int s = 0; s < 4; ++s) {
    const int bcol = n0 + s * 16 + l15;
    const float bb = bias ? bias[bcol] : 0.0f;
    #pragma unroll
    for (int i = 0; i < 8; ++i) {
      const int  mrow = tmRow + i + (halfId ? 8 : 0);
      const float val = accs[s][i] + bb;
      const size_t idx = (size_t)mrow * N + bcol;
      if (mode == 0)      outF[idx] = val;
      else if (mode == 1) outF[idx] = tanhf(val) * MAXOFF;
      else if (mode == 2) outH[idx] = (_Float16)val;
      else                outF[idx] = rowMask[mrow] ? val : 0.0f;
    }
  }
}

// ----------------------------- deformable sampler -----------------------------
// One wave32 per (b,t,h). Lane handles dim pair (d, d+32) -> natural RoPE pairing.

__global__ __launch_bounds__(256)
void sampler_k(const float* __restrict__ q, const float* __restrict__ off,
               const _Float16* __restrict__ k0p, const _Float16* __restrict__ k1p,
               const _Float16* __restrict__ k2p,
               const _Float16* __restrict__ v0p, const _Float16* __restrict__ v1p,
               const _Float16* __restrict__ v2p,
               const int* __restrict__ m0p, const int* __restrict__ m1p,
               const int* __restrict__ m2p,
               const unsigned char* __restrict__ mask, _Float16* __restrict__ attn) {
  const int lane = threadIdx.x & 31;
  const int w = (blockIdx.x * blockDim.x + threadIdx.x) >> 5;
  if (w >= BB * TT * NH) return;
  const int h = w % NH;
  const int t = (w / NH) % TT;
  const int b = w / (NH * TT);
  const int row = b * TT + t;

  const float invf = powf(10000.0f, -(float)lane / 32.0f);   // freq index = lane (0..31)

  // q + RoPE(pos = t)
  const float* qp = q + (size_t)row * DIMC + h * DHD;
  const float q1 = qp[lane], q2 = qp[lane + 32];
  const float angq = (float)t * invf;
  const float cq = cosf(angq), sq = sinf(angq);
  const float qc1 = q1 * cq - q2 * sq;
  const float qc2 = q1 * sq + q2 * cq;

  const _Float16* kps[NL] = {k0p, k1p, k2p};
  const _Float16* vps[NL] = {v0p, v1p, v2p};
  const int*      mps[NL] = {m0p, m1p, m2p};
  const int       tss[NL] = {TT, TT / 2, TT / 4};

  const float ref = (float)t / (float)(TT - 1);
  const int rowOK = mask[row] ? 1 : 0;

  float logits[NL * NKP], vs1[NL * NKP], vs2[NL * NKP];

  #pragma unroll
  for (int li = 0; li < NL; ++li) {
    const int Ts = tss[li];
    const _Float16* kp = kps[li];
    const _Float16* vp = vps[li];
    const int* mp = mps[li];
    #pragma unroll
    for (int kk = 0; kk < NKP; ++kk) {
      const int si = li * NKP + kk;
      const float o = off[(size_t)row * (NH * NL * NKP) + h * (NL * NKP) + si];
      const float s = fminf(fmaxf(ref + o, 0.0f), 1.0f);
      float idx = fminf(s * (float)(Ts - 1), (float)(Ts - 1) - 1e-6f);
      const int i0 = (int)floorf(idx);
      const int i1 = min(i0 + 1, Ts - 1);
      const float w1 = idx - (float)i0;
      const float w0 = 1.0f - w1;
      const int valid = mp[b * Ts + i0] | mp[b * Ts + i1];
      const size_t b0 = ((size_t)(b * Ts + i0)) * DIMC + h * DHD;
      const size_t b1 = ((size_t)(b * Ts + i1)) * DIMC + h * DHD;
      const float ks1 = w0 * (float)kp[b0 + lane]      + w1 * (float)kp[b1 + lane];
      const float ks2 = w0 * (float)kp[b0 + lane + 32] + w1 * (float)kp[b1 + lane + 32];
      vs1[si] = w0 * (float)vp[b0 + lane]      + w1 * (float)vp[b1 + lane];
      vs2[si] = w0 * (float)vp[b0 + lane + 32] + w1 * (float)vp[b1 + lane + 32];
      // RoPE on sampled k at pos = idx
      const float c2 = cosf(idx * invf), s2 = sinf(idx * invf);
      const float kr1 = ks1 * c2 - ks2 * s2;
      const float kr2 = ks1 * s2 + ks2 * c2;
      float part = qc1 * kr1 + qc2 * kr2;
      #pragma unroll
      for (int mm = 16; mm >= 1; mm >>= 1) part += __shfl_xor(part, mm, 32);
      logits[si] = (valid && rowOK) ? part * 0.125f : -1.0e9f;
    }
  }

  float mx = logits[0];
  #pragma unroll
  for (int i = 1; i < NL * NKP; ++i) mx = fmaxf(mx, logits[i]);
  float se = 0.0f, ex[NL * NKP];
  #pragma unroll
  for (int i = 0; i < NL * NKP; ++i) { ex[i] = expf(logits[i] - mx); se += ex[i]; }
  const float inv = 1.0f / se;
  float a1 = 0.0f, a2 = 0.0f;
  #pragma unroll
  for (int i = 0; i < NL * NKP; ++i) {
    const float a = ex[i] * inv;
    a1 += a * vs1[i];
    a2 += a * vs2[i];
  }
  _Float16* op = attn + (size_t)row * DIMC + h * DHD;
  op[lane]      = (_Float16)a1;
  op[lane + 32] = (_Float16)a2;
}

// ----------------------------- launch -----------------------------

extern "C" void kernel_launch(void* const* d_in, const int* in_sizes, int n_in,
                              void* d_out, int out_size, void* d_ws, size_t ws_size,
                              hipStream_t stream) {
  (void)in_sizes; (void)n_in; (void)out_size; (void)ws_size;
  const float*         x    = (const float*)d_in[0];
  const unsigned char* mask = (const unsigned char*)d_in[1];
  const float* Wq   = (const float*)d_in[2];
  const float* bq   = (const float*)d_in[3];
  const float* Wk   = (const float*)d_in[4];
  const float* bk   = (const float*)d_in[5];
  const float* Wv   = (const float*)d_in[6];
  const float* bv   = (const float*)d_in[7];
  const float* Woff = (const float*)d_in[8];
  const float* boff = (const float*)d_in[9];
  const float* Wo   = (const float*)d_in[10];
  const float* bo   = (const float*)d_in[11];
  float* out = (float*)d_out;

  const int M0 = BB * TT;         // 4096
  const int M1 = BB * TT / 2;     // 2048
  const int M2 = BB * TT / 4;     // 1024
  const int NOFF = NH * NL * NKP; // 192

  char* p = (char*)d_ws;
  auto alloc = [&](size_t bytes) -> void* {
    void* r = (void*)p;
    p += (bytes + 255) & ~(size_t)255;
    return r;
  };
  _Float16* xh    = (_Float16*)alloc((size_t)M0 * DIMC * 2);
  _Float16* Wqt   = (_Float16*)alloc((size_t)DIMC * DIMC * 2);
  _Float16* Wkt   = (_Float16*)alloc((size_t)DIMC * DIMC * 2);
  _Float16* Wvt   = (_Float16*)alloc((size_t)DIMC * DIMC * 2);
  _Float16* Wot   = (_Float16*)alloc((size_t)DIMC * DIMC * 2);
  _Float16* Wofft = (_Float16*)alloc((size_t)NOFF * DIMC * 2);
  float*    f1    = (float*)alloc((size_t)M1 * DIMC * 4);
  float*    f2    = (float*)alloc((size_t)M2 * DIMC * 4);
  _Float16* f1h   = (_Float16*)alloc((size_t)M1 * DIMC * 2);
  _Float16* f2h   = (_Float16*)alloc((size_t)M2 * DIMC * 2);
  int*      m0i   = (int*)alloc((size_t)M0 * 4);
  int*      m1i   = (int*)alloc((size_t)M1 * 4);
  int*      m2i   = (int*)alloc((size_t)M2 * 4);
  float*    qbuf  = (float*)alloc((size_t)M0 * DIMC * 4);
  float*    obuf  = (float*)alloc((size_t)M0 * NOFF * 4);
  _Float16* k0h   = (_Float16*)alloc((size_t)M0 * DIMC * 2);
  _Float16* k1h   = (_Float16*)alloc((size_t)M1 * DIMC * 2);
  _Float16* k2h   = (_Float16*)alloc((size_t)M2 * DIMC * 2);
  _Float16* v0h   = (_Float16*)alloc((size_t)M0 * DIMC * 2);
  _Float16* v1h   = (_Float16*)alloc((size_t)M1 * DIMC * 2);
  _Float16* v2h   = (_Float16*)alloc((size_t)M2 * DIMC * 2);
  _Float16* atnh  = (_Float16*)alloc((size_t)M0 * DIMC * 2);

  auto blocks = [](int n) { return (n + 255) / 256; };

  // conversions / transposes
  cvt_f32_to_f16_k<<<blocks(M0 * DIMC), 256, 0, stream>>>(x, xh, M0 * DIMC);
  transpose_cvt_k<<<blocks(DIMC * DIMC), 256, 0, stream>>>(Wq, Wqt, DIMC, DIMC);
  transpose_cvt_k<<<blocks(DIMC * DIMC), 256, 0, stream>>>(Wk, Wkt, DIMC, DIMC);
  transpose_cvt_k<<<blocks(DIMC * DIMC), 256, 0, stream>>>(Wv, Wvt, DIMC, DIMC);
  transpose_cvt_k<<<blocks(DIMC * DIMC), 256, 0, stream>>>(Wo, Wot, DIMC, DIMC);
  transpose_cvt_k<<<blocks(DIMC * NOFF), 256, 0, stream>>>(Woff, Wofft, DIMC, NOFF);
  mask_to_int_k<<<blocks(M0), 256, 0, stream>>>(mask, m0i, M0);

  // feature pyramid
  downsample_k<<<blocks(M1 * DIMC), 256, 0, stream>>>(x, m0i, f1, m1i, TT, TT / 2);
  downsample_k<<<blocks(M2 * DIMC), 256, 0, stream>>>(f1, m1i, f2, m2i, TT / 2, TT / 4);
  cvt_f32_to_f16_k<<<blocks(M1 * DIMC), 256, 0, stream>>>(f1, f1h, M1 * DIMC);
  cvt_f32_to_f16_k<<<blocks(M2 * DIMC), 256, 0, stream>>>(f2, f2h, M2 * DIMC);

  auto gemm = [&](const _Float16* A, const _Float16* Bt, const float* bias,
                  float* oF, _Float16* oH, const unsigned char* rm,
                  int M, int N, int mode) {
    int nblocks = (M >> 7) * (N >> 6);   // 128x64 block tiles; all dims divide exactly
    gemm_wmma_f16_k<<<nblocks, 256, 0, stream>>>(A, Bt, bias, oF, oH, rm,
                                                 M, N, DIMC, mode);
  };

  // projections
  gemm(xh, Wqt, bq,     qbuf, nullptr, nullptr, M0, DIMC, 0);
  gemm(xh, Wofft, boff, obuf, nullptr, nullptr, M0, NOFF, 1);
  gemm(xh,  Wkt, bk, nullptr, k0h, nullptr, M0, DIMC, 2);
  gemm(xh,  Wvt, bv, nullptr, v0h, nullptr, M0, DIMC, 2);
  gemm(f1h, Wkt, bk, nullptr, k1h, nullptr, M1, DIMC, 2);
  gemm(f1h, Wvt, bv, nullptr, v1h, nullptr, M1, DIMC, 2);
  gemm(f2h, Wkt, bk, nullptr, k2h, nullptr, M2, DIMC, 2);
  gemm(f2h, Wvt, bv, nullptr, v2h, nullptr, M2, DIMC, 2);

  // deformable sampling + attention
  sampler_k<<<blocks(BB * TT * NH * 32), 256, 0, stream>>>(
      qbuf, obuf, k0h, k1h, k2h, v0h, v1h, v2h, m0i, m1i, m2i, mask, atnh);

  // output projection with mask
  gemm(atnh, Wot, bo, out, nullptr, mask, M0, DIMC, 3);
}
